// GBDT2NN_65274912965364
// MI455X (gfx1250) — matrix operand order repaired and checked by reading, not compiled, not run on hardware
//
#include <hip/hip_runtime.h>
#include <hip/hip_bf16.h>

// ---------------------------------------------------------------------------
// GBDT2NN fused pipeline for gfx1250 (MI455X), bf16 WMMA + fp32 accumulate.
// Native __bf16 conversions so hardware v_cvt_* is used instead of bit tricks.
// ---------------------------------------------------------------------------

typedef __attribute__((ext_vector_type(16))) __bf16 bf16x16;
typedef __attribute__((ext_vector_type(8)))  __bf16 bf16x8;
typedef __attribute__((ext_vector_type(8)))  float  f32x8;

#define BATCH    8192
#define NMODELS  128
#define NFEAT    32
#define L1       64
#define L2DIM    32
#define L3       16
#define INSIZE   512
#define PREDW    (NMODELS * L3)   // 2048

__device__ __forceinline__ f32x8 wmma_bf16(bf16x16 a, bf16x16 b, f32x8 c) {
  return __builtin_amdgcn_wmma_f32_16x16x32_bf16(
      false, a, false, b, (short)0, c, false, false);
}

// ---------------------------------------------------------------------------
// Utility kernels
// ---------------------------------------------------------------------------
__global__ void zero_kernel(float* __restrict__ p, int n) {
  int i = blockIdx.x * 256 + threadIdx.x;
  if (i < n) p[i] = 0.0f;
}

// W: [128][K][N] fp32  ->  Wt: [128][N][K] bf16 (column-major for B-fragments)
__global__ void prep_w_kernel(const float* __restrict__ W,
                              __bf16* __restrict__ Wt, int K, int N) {
  int idx = blockIdx.x * 256 + threadIdx.x;
  int total = NMODELS * K * N;
  if (idx >= total) return;
  int m   = idx / (K * N);
  int rem = idx - m * K * N;
  int k   = rem / N;
  int n   = rem - k * N;
  Wt[((size_t)m * N + n) * K + k] = (__bf16)W[idx];
}

// scale = gamma * rsqrt(var + eps), shift = beta - mean*scale
__global__ void bnstats_kernel(const float* __restrict__ sum, const float* __restrict__ sq,
                               const float* __restrict__ gamma, const float* __restrict__ beta,
                               float* __restrict__ scale, float* __restrict__ shift, int F) {
  int f = blockIdx.x * 256 + threadIdx.x;
  if (f >= F) return;
  const float invB = 1.0f / (float)BATCH;
  float mean = sum[f] * invB;
  float var  = sq[f] * invB - mean * mean;
  float sc   = gamma[f] * rsqrtf(var + 1e-5f);
  scale[f] = sc;
  shift[f] = beta[f] - mean * sc;
}

// ---------------------------------------------------------------------------
// Layer 1: gather(x) [16x32] x W1 [32x64] -> z1 bf16 [m][8192][64] + BN stats
// grid (64,128), 256 threads (8 waves x 16 rows)
// ---------------------------------------------------------------------------
__global__ __launch_bounds__(256) void gemm1_kernel(
    const float* __restrict__ x, const int* __restrict__ uf,
    const __bf16* __restrict__ W1t, const float* __restrict__ b1,
    __bf16* __restrict__ z1,
    float* __restrict__ sum1, float* __restrict__ sq1) {
  __shared__ float s_sum[L1];
  __shared__ float s_sq[L1];
  const int m    = blockIdx.y;
  const int wave = threadIdx.x >> 5;
  const int lane = threadIdx.x & 31;
  const int r    = lane & 15;
  const int kg   = lane >> 4;
  const int row0 = blockIdx.x * 128 + wave * 16;

  if (threadIdx.x < L1) { s_sum[threadIdx.x] = 0.f; s_sq[threadIdx.x] = 0.f; }
  __syncthreads();

  // ---- A fragment: 16x32 bf16 gathered from x -------------------------------
  const int* ufm = uf + m * NFEAT;
  int idxs[16];
  *(int4*)&idxs[0]  = *(const int4*)(ufm + kg * 16);
  *(int4*)&idxs[4]  = *(const int4*)(ufm + kg * 16 + 4);
  *(int4*)&idxs[8]  = *(const int4*)(ufm + kg * 16 + 8);
  *(int4*)&idxs[12] = *(const int4*)(ufm + kg * 16 + 12);
  const float* xrow = x + (size_t)(row0 + r) * INSIZE;
  bf16x16 a;
#pragma unroll
  for (int e = 0; e < 8; ++e) {
    a[e]     = (__bf16)xrow[idxs[e]];       // K = kg*16 + e
    a[e + 8] = (__bf16)xrow[idxs[8 + e]];   // K = kg*16 + 8 + e
  }

#pragma unroll
  for (int ct = 0; ct < 4; ++ct) {
    const int n = ct * 16 + r;
    const bf16x16 b = *(const bf16x16*)(W1t + ((size_t)(m * L1 + n) * NFEAT + kg * 16));
    const float bias = b1[m * L1 + n];
    f32x8 acc;
#pragma unroll
    for (int v = 0; v < 8; ++v) acc[v] = bias;
    acc = wmma_bf16(a, b, acc);

    float s = 0.f, q = 0.f;
    __bf16* zp = z1 + ((size_t)m * BATCH + row0) * L1 + ct * 16 + r;
#pragma unroll
    for (int v = 0; v < 8; ++v) {
      const float c = acc[v];                    // row = row0 + v + kg*8
      s += c; q += c * c;
      zp[(size_t)(v + kg * 8) * L1] = (__bf16)c;
    }
    atomicAdd(&s_sum[ct * 16 + r], s);
    atomicAdd(&s_sq [ct * 16 + r], q);
  }
  __syncthreads();
  if (threadIdx.x < L1) {
    atomicAdd(&sum1[m * L1 + threadIdx.x], s_sum[threadIdx.x]);
    atomicAdd(&sq1 [m * L1 + threadIdx.x], s_sq [threadIdx.x]);
  }
}

// ---------------------------------------------------------------------------
// Layer 2: relu(BN1(z1)) [16x64] x W2 [64x32] -> z2 bf16 [m][8192][32] + stats
// ---------------------------------------------------------------------------
__global__ __launch_bounds__(256) void gemm2_kernel(
    const __bf16* __restrict__ z1, const __bf16* __restrict__ W2t,
    const float* __restrict__ b2,
    const float* __restrict__ scale1, const float* __restrict__ shift1,
    __bf16* __restrict__ z2,
    float* __restrict__ sum2, float* __restrict__ sq2) {
  __shared__ float s_sum[L2DIM];
  __shared__ float s_sq[L2DIM];
  const int m    = blockIdx.y;
  const int wave = threadIdx.x >> 5;
  const int lane = threadIdx.x & 31;
  const int r    = lane & 15;
  const int kg   = lane >> 4;
  const int row0 = blockIdx.x * 128 + wave * 16;

  if (threadIdx.x < L2DIM) { s_sum[threadIdx.x] = 0.f; s_sq[threadIdx.x] = 0.f; }
  __syncthreads();

  f32x8 acc[2];
#pragma unroll
  for (int ct = 0; ct < 2; ++ct) {
    const float bias = b2[m * L2DIM + ct * 16 + r];
#pragma unroll
    for (int v = 0; v < 8; ++v) acc[ct][v] = bias;
  }

  const __bf16* zrow = z1 + ((size_t)m * BATCH + row0 + r) * L1;
#pragma unroll
  for (int kt = 0; kt < 2; ++kt) {
    const int k0 = kt * 32;
    const bf16x8 c0 = *(const bf16x8*)(zrow + k0 + kg * 16);       // K = k0 + kg*16 ..
    const bf16x8 c1 = *(const bf16x8*)(zrow + k0 + kg * 16 + 8);   // K = k0+kg*16+8 ..
    const float* sc = scale1 + m * L1 + k0 + kg * 16;
    const float* sh = shift1 + m * L1 + k0 + kg * 16;
    bf16x16 af;
#pragma unroll
    for (int e = 0; e < 8; ++e) {
      float v0 = (float)c0[e] * sc[e]      + sh[e];
      float v1 = (float)c1[e] * sc[e + 8] + sh[e + 8];
      af[e]     = (__bf16)fmaxf(v0, 0.f);
      af[e + 8] = (__bf16)fmaxf(v1, 0.f);
    }
#pragma unroll
    for (int ct = 0; ct < 2; ++ct) {
      const int n = ct * 16 + r;
      const bf16x16 b = *(const bf16x16*)(W2t + ((size_t)(m * L2DIM + n) * L1 + k0 + kg * 16));
      acc[ct] = wmma_bf16(af, b, acc[ct]);
    }
  }

#pragma unroll
  for (int ct = 0; ct < 2; ++ct) {
    float s = 0.f, q = 0.f;
    __bf16* zp = z2 + ((size_t)m * BATCH + row0) * L2DIM + ct * 16 + r;
#pragma unroll
    for (int v = 0; v < 8; ++v) {
      const float c = acc[ct][v];
      s += c; q += c * c;
      zp[(size_t)(v + kg * 8) * L2DIM] = (__bf16)c;
    }
    atomicAdd(&s_sum[ct * 16 + r], s);
    atomicAdd(&s_sq [ct * 16 + r], q);
  }
  __syncthreads();
  if (threadIdx.x < L2DIM) {
    atomicAdd(&sum2[m * L2DIM + threadIdx.x], s_sum[threadIdx.x]);
    atomicAdd(&sq2 [m * L2DIM + threadIdx.x], s_sq [threadIdx.x]);
  }
}

// ---------------------------------------------------------------------------
// Layer 3: relu(BN2(z2)) [16x32] x W3 [32x16] -> pred fp32 [8192][2048]
// ---------------------------------------------------------------------------
__global__ __launch_bounds__(256) void gemm3_kernel(
    const __bf16* __restrict__ z2, const __bf16* __restrict__ W3t,
    const float* __restrict__ b3,
    const float* __restrict__ scale2, const float* __restrict__ shift2,
    float* __restrict__ pred) {
  const int m    = blockIdx.y;
  const int wave = threadIdx.x >> 5;
  const int lane = threadIdx.x & 31;
  const int r    = lane & 15;
  const int kg   = lane >> 4;
  const int row0 = blockIdx.x * 128 + wave * 16;

  const __bf16* zrow = z2 + ((size_t)m * BATCH + row0 + r) * L2DIM;
  const bf16x8 c0 = *(const bf16x8*)(zrow + kg * 16);
  const bf16x8 c1 = *(const bf16x8*)(zrow + kg * 16 + 8);
  const float* sc = scale2 + m * L2DIM + kg * 16;
  const float* sh = shift2 + m * L2DIM + kg * 16;
  bf16x16 af;
#pragma unroll
  for (int e = 0; e < 8; ++e) {
    float v0 = (float)c0[e] * sc[e]      + sh[e];
    float v1 = (float)c1[e] * sc[e + 8] + sh[e + 8];
    af[e]     = (__bf16)fmaxf(v0, 0.f);
    af[e + 8] = (__bf16)fmaxf(v1, 0.f);
  }

  const int n = r;
  const bf16x16 b = *(const bf16x16*)(W3t + ((size_t)(m * L3 + n) * L2DIM + kg * 16));
  const float bias = b3[m * L3 + n];
  f32x8 acc;
#pragma unroll
  for (int v = 0; v < 8; ++v) acc[v] = bias;
  acc = wmma_bf16(af, b, acc);

  float* pp = pred + (size_t)row0 * PREDW + m * L3 + n;
#pragma unroll
  for (int v = 0; v < 8; ++v)
    pp[(size_t)(v + kg * 8) * PREDW] = acc[v];
}

// ---------------------------------------------------------------------------
// out[b] = pred[b,:] . out_weight + out_bias   (one wave per row)
// ---------------------------------------------------------------------------
__global__ __launch_bounds__(256) void out_kernel(
    const float* __restrict__ pred, const float* __restrict__ ow,
    const float* __restrict__ ob, float* __restrict__ out) {
  const int row  = blockIdx.x * 8 + (threadIdx.x >> 5);
  const int lane = threadIdx.x & 31;
  const float4* p = (const float4*)(pred + (size_t)row * PREDW);
  const float4* w = (const float4*)ow;
  float s = 0.f;
#pragma unroll
  for (int j = 0; j < 16; ++j) {
    float4 a = p[lane + 32 * j];
    float4 b = w[lane + 32 * j];
    s += a.x * b.x + a.y * b.y + a.z * b.z + a.w * b.w;
  }
#pragma unroll
  for (int off = 16; off > 0; off >>= 1) s += __shfl_xor(s, off, 32);
  if (lane == 0) out[row] = s + ob[0];
}

// ---------------------------------------------------------------------------
extern "C" void kernel_launch(void* const* d_in, const int* in_sizes, int n_in,
                              void* d_out, int out_size, void* d_ws, size_t ws_size,
                              hipStream_t stream) {
  const float* x   = (const float*)d_in[0];
  const int*   uf  = (const int*)  d_in[1];
  const float* W1  = (const float*)d_in[2];
  const float* b1  = (const float*)d_in[3];
  const float* W2  = (const float*)d_in[4];
  const float* b2  = (const float*)d_in[5];
  const float* W3  = (const float*)d_in[6];
  const float* b3  = (const float*)d_in[7];
  const float* g1  = (const float*)d_in[8];
  const float* be1 = (const float*)d_in[9];
  const float* g2  = (const float*)d_in[10];
  const float* be2 = (const float*)d_in[11];
  const float* ow  = (const float*)d_in[12];
  const float* ob  = (const float*)d_in[13];

  float* out  = (float*)d_out;       // [8192]
  float* pred = out + BATCH;         // [8192][2048]

  char* w = (char*)d_ws;
  auto take = [&](size_t bytes) {
    void* p = (void*)w;
    w += (bytes + 255) & ~(size_t)255;
    return p;
  };
  __bf16* z1  = (__bf16*)take((size_t)NMODELS * BATCH * L1 * 2);
  __bf16* z2  = (__bf16*)take((size_t)NMODELS * BATCH * L2DIM * 2);
  __bf16* W1t = (__bf16*)take((size_t)NMODELS * L1 * NFEAT * 2);
  __bf16* W2t = (__bf16*)take((size_t)NMODELS * L2DIM * L1 * 2);
  __bf16* W3t = (__bf16*)take((size_t)NMODELS * L3 * L2DIM * 2);
  float* sum1   = (float*)take((size_t)NMODELS * L1 * 4);
  float* sq1    = (float*)take((size_t)NMODELS * L1 * 4);
  float* sum2   = (float*)take((size_t)NMODELS * L2DIM * 4);
  float* sq2    = (float*)take((size_t)NMODELS * L2DIM * 4);
  float* scale1 = (float*)take((size_t)NMODELS * L1 * 4);
  float* shift1 = (float*)take((size_t)NMODELS * L1 * 4);
  float* scale2 = (float*)take((size_t)NMODELS * L2DIM * 4);
  float* shift2 = (float*)take((size_t)NMODELS * L2DIM * 4);

  // sum1/sq1/sum2/sq2 are contiguous (each a multiple of 256 bytes): zero all.
  const int nstats = NMODELS * L1 * 2 + NMODELS * L2DIM * 2;   // 24576
  zero_kernel<<<(nstats + 255) / 256, 256, 0, stream>>>(sum1, nstats);

  prep_w_kernel<<<(NMODELS * NFEAT * L1 + 255) / 256, 256, 0, stream>>>(W1, W1t, NFEAT, L1);
  prep_w_kernel<<<(NMODELS * L1 * L2DIM + 255) / 256, 256, 0, stream>>>(W2, W2t, L1, L2DIM);
  prep_w_kernel<<<(NMODELS * L2DIM * L3 + 255) / 256, 256, 0, stream>>>(W3, W3t, L2DIM, L3);

  gemm1_kernel<<<dim3(BATCH / 128, NMODELS), 256, 0, stream>>>(x, uf, W1t, b1, z1, sum1, sq1);
  bnstats_kernel<<<(NMODELS * L1 + 255) / 256, 256, 0, stream>>>(sum1, sq1, g1, be1, scale1, shift1, NMODELS * L1);
  gemm2_kernel<<<dim3(BATCH / 128, NMODELS), 256, 0, stream>>>(z1, W2t, b2, scale1, shift1, z2, sum2, sq2);
  bnstats_kernel<<<(NMODELS * L2DIM + 255) / 256, 256, 0, stream>>>(sum2, sq2, g2, be2, scale2, shift2, NMODELS * L2DIM);
  gemm3_kernel<<<dim3(BATCH / 128, NMODELS), 256, 0, stream>>>(z2, W3t, b3, scale2, shift2, pred);
  out_kernel<<<BATCH / 8, 256, 0, stream>>>(pred, ow, ob, out);
}
